// AMMaskedLinear_51934744543574
// MI455X (gfx1250) — matrix operand order repaired and checked by reading, not compiled, not run on hardware
//
#include <hip/hip_runtime.h>
#include <hip/hip_bf16.h>

// ---------------------------------------------------------------------------
// AMMaskedLinear on MI455X (gfx1250, wave32, WMMA + async global->LDS)
//
// Shapes: B=64, IN=1024, OUT=1024, D=32.
// presence bitmask per batch -> expand x into X'[2048 x 1024] (one row per
// (batch, threshold t=1..32)) -> single bf16 NT-GEMM via
// v_wmma_f32_16x16x32_bf16 with double-buffered GLOBAL_LOAD_ASYNC_TO_LDS
// staging -> fused gather epilogue:
// out[b,k] = om * (cscale_b[k] * Y'[b*32 + mhigh-1, k] + cbias_b[k]).
// ---------------------------------------------------------------------------

typedef __attribute__((ext_vector_type(16))) __bf16 v16bf;
typedef __attribute__((ext_vector_type(8)))  float  v8f;

static constexpr int BATCH = 64;
static constexpr int IN    = 1024;
static constexpr int OUT   = 1024;
static constexpr int MEXP  = BATCH * 32;   // 2048 expanded rows

// workspace layout (bytes)
static constexpr size_t WS_PRES = 0;                        // 64 * u64      = 512
static constexpr size_t WS_MH   = 4096;                     // 64*1024 * i32 = 256K
static constexpr size_t WS_XP   = 266240;                   // 2048*1024 bf16 = 4M
static constexpr size_t WS_DB   = 4460544;                  // 1024*1024 bf16 = 2M

__device__ __forceinline__ unsigned short f2bf(float f) {
  unsigned int u = __builtin_bit_cast(unsigned int, f);
  unsigned int r = u + 0x7FFFu + ((u >> 16) & 1u);   // round-to-nearest-even
  return (unsigned short)(r >> 16);
}

// Generic pointer to a __shared__ object is {shared_aperture_hi32, lds_offset};
// the low 32 bits are exactly the LDS byte address the async-DMA ops expect.
__device__ __forceinline__ unsigned lds_addr32(const void* p) {
  return (unsigned)(unsigned long long)(uintptr_t)p;
}

// ---- kernel 1: 33-bit presence bitmask per batch --------------------------
__global__ __launch_bounds__(256) void presence_kernel(
    const int* __restrict__ hidden_rank, unsigned long long* __restrict__ pres) {
  __shared__ unsigned int s_lo, s_hi;
  const int b = blockIdx.x;
  if (threadIdx.x == 0) { s_lo = 0u; s_hi = 0u; }
  __syncthreads();
  unsigned int mylo = 0u, myhi = 0u;
#pragma unroll
  for (int e = 0; e < 4; ++e) {
    const int v = hidden_rank[b * IN + threadIdx.x * 4 + e] & 63;
    if (v < 32) mylo |= (1u << v);
    else        myhi |= (1u << (v - 32));
  }
  if (mylo) atomicOr(&s_lo, mylo);
  if (myhi) atomicOr(&s_hi, myhi);
  __syncthreads();
  if (threadIdx.x == 0)
    pres[b] = ((unsigned long long)s_hi << 32) | (unsigned long long)s_lo;
}

// ---- kernel 2: mhigh per (b,k) and out init = om * cbias_b ---------------
__global__ __launch_bounds__(256) void prep_kernel(
    const int* __restrict__ r_high, const float* __restrict__ cbias_b,
    const unsigned long long* __restrict__ pres,
    int* __restrict__ mh, float* __restrict__ out) {
  const int idx = blockIdx.x * 256 + threadIdx.x;   // over B*OUT = 65536
  const int b = idx >> 10;
  const int k = idx & 1023;
  const int rh = r_high[k] & 63;
  const bool om = (pres[b] >> rh) & 1ull;
  mh[idx]  = om ? rh : 0;
  out[idx] = om ? cbias_b[k] : 0.0f;
}

// ---- kernel 3: build expanded bf16 A-matrix X'[2048 x 1024] ---------------
__global__ __launch_bounds__(256) void pack_x_kernel(
    const float* __restrict__ x, const int* __restrict__ r_low,
    const unsigned long long* __restrict__ pres, unsigned short* __restrict__ Xp) {
  const int base = (blockIdx.x * 256 + threadIdx.x) * 8;  // over 2048*1024
  const int m  = base >> 10;
  const int j0 = base & 1023;
  const int b  = m >> 5;
  const int t  = (m & 31) + 1;
  const unsigned long long p = pres[b];
  union { unsigned short v[8]; uint4 q; } o;
#pragma unroll
  for (int e = 0; e < 8; ++e) {
    const int j  = j0 + e;
    const int rl = r_low[j] & 63;
    const int mlow = ((p >> rl) & 1ull) ? rl : 0;
    const float val = (mlow >= 1 && mlow <= t) ? x[b * IN + j] : 0.0f;
    o.v[e] = f2bf(val);
  }
  *(uint4*)(Xp + base) = o.q;
}

// ---- kernel 4: direction f32 -> bf16 --------------------------------------
__global__ __launch_bounds__(256) void cvt_dir_kernel(
    const float* __restrict__ dir, unsigned short* __restrict__ Db) {
  const int i0 = (blockIdx.x * 256 + threadIdx.x) * 4;    // over 1024*1024
  const float4 f = *(const float4*)(dir + i0);
  uint2 o;
  o.x = (unsigned)f2bf(f.x) | ((unsigned)f2bf(f.y) << 16);
  o.y = (unsigned)f2bf(f.z) | ((unsigned)f2bf(f.w) << 16);
  *(uint2*)(Db + i0) = o;
}

// ---- kernel 5: bf16 NT-GEMM (2048 x 1024 x 1024) with fused gather --------
// Block tile 128(M) x 64(N); 8 waves: 4 along M x 2 along N, each wave 32x32.
// Double-buffered async global->LDS staging (3 async b128 ops per thread/tile).
__global__ __launch_bounds__(256) void gemm_kernel(
    const unsigned short* __restrict__ Xp, const unsigned short* __restrict__ Db,
    const int* __restrict__ mh, const float* __restrict__ cscale_b,
    const float* __restrict__ cbias_b, float* __restrict__ out) {
  constexpr int K = IN;
  constexpr int LDA = 40;  // padded LDS row stride (halves), 80B keeps 16B align
  __shared__ alignas(16) unsigned short As[2][128 * LDA];
  __shared__ alignas(16) unsigned short Bs[2][64 * LDA];

  const int tid   = threadIdx.x;
  const int lane  = tid & 31;
  const int wid   = tid >> 5;
  const int waveM = wid & 3;    // 0..3
  const int waveN = wid >> 2;   // 0..1
  const int bM = blockIdx.y * 128;
  const int bN = blockIdx.x * 64;

  const int hi  = (lane >> 4) & 1;  // lane >= 16
  const int r15 = lane & 15;

  const int arow = tid >> 1;            // 0..127
  const int ah   = (tid & 1) * 16;      // half-row offset (halves)
  const int brow = tid >> 2;            // 0..63
  const int bh   = (tid & 3) * 8;       // quarter-row offset (halves)

  // Issue one K-tile's staging: A 128x32 (2 async b128/thread), B 64x32 (1).
  auto issue_tile = [&](int buf, int kk) {
    const unsigned short* gA = Xp + (size_t)(bM + arow) * K + kk + ah;
    const unsigned short* gB = Db + (size_t)(bN + brow) * K + kk + bh;
    const unsigned lA = lds_addr32(&As[buf][arow * LDA + ah]);
    const unsigned lB = lds_addr32(&Bs[buf][brow * LDA + bh]);
    // IOFFSET applies to both the global and LDS address (ISA 15.18.3).
    asm volatile("global_load_async_to_lds_b128 %0, %1, off"
                 :: "v"(lA), "v"(gA) : "memory");
    asm volatile("global_load_async_to_lds_b128 %0, %1, off offset:16"
                 :: "v"(lA), "v"(gA) : "memory");
    asm volatile("global_load_async_to_lds_b128 %0, %1, off"
                 :: "v"(lB), "v"(gB) : "memory");
  };

  v8f acc[2][2] = {};

  issue_tile(0, 0);
  int cur = 0;
  for (int kk = 0; kk < K; kk += 32) {
    if (kk + 32 < K) {
      issue_tile(cur ^ 1, kk + 32);                 // next tile in flight
      asm volatile("s_wait_asynccnt 0x3" ::: "memory");  // current tile done
    } else {
      asm volatile("s_wait_asynccnt 0x0" ::: "memory");
    }
    if (kk + 64 < K)  // pull A two tiles ahead into cache
      __builtin_prefetch(Xp + (size_t)(bM + arow) * K + kk + 64 + ah, 0, 3);
    __syncthreads();

    // Per-lane fragment gathers matching documented 16-bit layouts:
    // A 16x32: lane<16 holds K {0..7,16..23}, lane>=16 holds K {8..15,24..31}
    // B 32x16: lane<16 holds K {0..7,16..23}, lane>=16 holds K {8..15,24..31} (lane&15 = N)
    union Frag { uint4 q[2]; v16bf v; };
    Frag a[2], b[2];
    const int koff = hi * 8;
#pragma unroll
    for (int i = 0; i < 2; ++i) {
      const unsigned short* ap = &As[cur][(waveM * 32 + i * 16 + r15) * LDA];
      a[i].q[0] = *(const uint4*)(ap + koff);
      a[i].q[1] = *(const uint4*)(ap + koff + 16);
    }
#pragma unroll
    for (int j = 0; j < 2; ++j) {
      const unsigned short* bp =
          &Bs[cur][(waveN * 32 + j * 16 + r15) * LDA] + hi * 8;
      b[j].q[0] = *(const uint4*)(bp);
      b[j].q[1] = *(const uint4*)(bp + 16);
    }
#pragma unroll
    for (int i = 0; i < 2; ++i)
#pragma unroll
      for (int j = 0; j < 2; ++j)
        acc[i][j] = __builtin_amdgcn_wmma_f32_16x16x32_bf16(
            false, a[i].v, false, b[j].v, (short)0, acc[i][j], false, false);
    __syncthreads();  // protect buffer 'cur' before it is refilled next+1 iter
    cur ^= 1;
  }

  // Fused gather epilogue: row m encodes (batch b = m>>5, threshold t = (m&31)+1).
  // Exactly one row matches mh[b][n], so the scatter is race-free.
#pragma unroll
  for (int i = 0; i < 2; ++i) {
#pragma unroll
    for (int j = 0; j < 2; ++j) {
      const int n = bN + waveN * 32 + j * 16 + r15;
#pragma unroll
      for (int v = 0; v < 8; ++v) {
        const int m = bM + waveM * 32 + i * 16 + v + hi * 8;
        const int bb = m >> 5;
        const int t  = (m & 31) + 1;
        if (mh[bb * OUT + n] == t)
          out[bb * OUT + n] = cscale_b[n] * acc[i][j][v] + cbias_b[n];
      }
    }
  }
}

extern "C" void kernel_launch(void* const* d_in, const int* in_sizes, int n_in,
                              void* d_out, int out_size, void* d_ws, size_t ws_size,
                              hipStream_t stream) {
  const float* x           = (const float*)d_in[0];
  // d_in[1] mask, d_in[2] pre_mask: unused by the reference
  const int*   hidden_rank = (const int*)d_in[3];
  const int*   r_low       = (const int*)d_in[4];
  const int*   r_high      = (const int*)d_in[5];
  const float* direction   = (const float*)d_in[6];
  // d_in[7] cscale_w, d_in[9] cbias_w: multiply zeros in the reference
  const float* cscale_b    = (const float*)d_in[8];
  const float* cbias_b     = (const float*)d_in[10];
  float* out = (float*)d_out;

  char* ws = (char*)d_ws;
  unsigned long long* pres = (unsigned long long*)(ws + WS_PRES);
  int*            mh = (int*)(ws + WS_MH);
  unsigned short* Xp = (unsigned short*)(ws + WS_XP);
  unsigned short* Db = (unsigned short*)(ws + WS_DB);

  presence_kernel<<<BATCH, 256, 0, stream>>>(hidden_rank, pres);
  prep_kernel<<<(BATCH * OUT) / 256, 256, 0, stream>>>(r_high, cbias_b, pres, mh, out);
  pack_x_kernel<<<(MEXP * IN) / (256 * 8), 256, 0, stream>>>(x, r_low, pres, Xp);
  cvt_dir_kernel<<<(OUT * IN) / (256 * 4), 256, 0, stream>>>(direction, Db);
  gemm_kernel<<<dim3(OUT / 64, MEXP / 128), 256, 0, stream>>>(Xp, Db, mh, cscale_b,
                                                              cbias_b, out);
}
